// HierachicalAttentionNetwork_18588618457133
// MI455X (gfx1250) — matrix operand order repaired and checked by reading, not compiled, not run on hardware
//
#include <hip/hip_runtime.h>
#include <math.h>

// ---------------------------------------------------------------------------
// HierarchicalAttentionNetwork forward for gfx1250 (MI455X), bf16 WMMA path.
// ---------------------------------------------------------------------------

typedef __bf16 bf16_t;
typedef __bf16 v16bf __attribute__((ext_vector_type(16)));
typedef __bf16 v8bf  __attribute__((ext_vector_type(8)));
typedef float  v8f   __attribute__((ext_vector_type(8)));

// Problem constants
constexpr int ED   = 200;   // embedding dim == 2*HW == 2*HS (GRU input dims)
constexpr int EDP  = 224;   // padded K (7 tiles of 32)
constexpr int KT   = 7;     // K tiles
constexpr int GN   = 600;   // gate GEMM N (3*H * 2 dirs)
constexpr int NTG  = 38;    // gate GEMM N tiles (608 padded)
constexpr int AN   = 200;   // attention GEMM N
constexpr int NTA  = 13;    // attention N tiles (208 padded)
constexpr int WW   = 50;    // words per sentence
constexpr int WMP  = 64;    // padded word rows (4 M tiles)
constexpr int SS   = 40;    // sentences per doc
constexpr int SMP  = 48;    // padded sentence rows (3 M tiles)
constexpr int BB   = 128;   // batch
constexpr int NC   = 10;
constexpr int FRAG = 512;   // bf16 elements per packed 32x16 B fragment

#define WMMA_BF16(A, Bm, C) \
  __builtin_amdgcn_wmma_f32_16x16x32_bf16(false, (A), false, (Bm), (short)0, (C), false, false)

__device__ __forceinline__ float sigf(float x) { return 1.f / (1.f + __expf(-x)); }

// Load A fragment (16x32 bf16, M x K) from row-major LDS [rows][EDP].
// ISA layout: lanes 0-15 -> M=lane, K = {0..7, 16..23}; lanes 16-31 -> K = {8..15, 24..31}.
__device__ __forceinline__ v16bf load_a(const bf16_t* base, int mt, int kt, int lane) {
  int row = mt * 16 + (lane & 15);
  int kb  = (lane >> 4) << 3;  // 0 or 8
  const bf16_t* p = base + row * EDP + kt * 32 + kb;
  v8bf lo = *(const v8bf*)p;
  v8bf hi = *(const v8bf*)(p + 16);
  v16bf a;
#pragma unroll
  for (int i = 0; i < 8; ++i) { a[i] = lo[i]; a[i + 8] = hi[i]; }
  return a;
}

// Load B fragment (32x16 bf16, K x N) from pre-packed global buffer:
// fragment-major, 16 contiguous bf16 per lane.
__device__ __forceinline__ v16bf load_b(const bf16_t* pack, int nt, int kt, int lane) {
  return *(const v16bf*)(pack + (((nt * KT + kt) << 9) + (lane << 4)));
}

// ---------------------------------------------------------------------------
// Pack a (possibly 2-part) weight matrix W (rows = output N, cols = K=200,
// GEMM computes X @ W.T) into WMMA B-fragment layout, fp32 -> bf16.
// B layout: lane L holds N = nt*16 + (L&15), K = kt*32 + (L>>4)*8 + (i&7) + (i>>3)*16.
// ---------------------------------------------------------------------------
__global__ void pack_b_kernel(const float* __restrict__ Wf, const float* __restrict__ Wb,
                              int rowsPerDir, int Nreal, int NT, bf16_t* __restrict__ dst) {
  int t = blockIdx.x * blockDim.x + threadIdx.x;
  int total = NT * KT * FRAG;
  if (t >= total) return;
  int frag = t >> 9;
  int r    = t & 511;
  int lane = r >> 4;
  int i    = r & 15;
  int nt = frag / KT;
  int kt = frag % KT;
  int n = nt * 16 + (lane & 15);
  int k = kt * 32 + ((lane >> 4) << 3) + (i & 7) + ((i >> 3) << 4);
  float v = 0.f;
  if (n < Nreal && k < ED) {
    v = (n < rowsPerDir) ? Wf[n * ED + k] : Wb[(n - rowsPerDir) * ED + k];
  }
  dst[t] = (bf16_t)v;
}

// ---------------------------------------------------------------------------
// Word-level kernel: one block per (b,s) sentence. 8 waves / 256 threads.
// embed gather -> bf16 LDS -> WMMA gate GEMM -> GRU gates -> WMMA attention
// GEMM -> softmax pooling -> svec[b,s,200].
// ---------------------------------------------------------------------------
__global__ __launch_bounds__(256) void word_kernel(
    const int* __restrict__ x, const float* __restrict__ emb,
    const float* __restrict__ bih_f, const float* __restrict__ bhh_f,
    const float* __restrict__ bih_b, const float* __restrict__ bhh_b,
    const bf16_t* __restrict__ Wpack, const bf16_t* __restrict__ WApack,
    const float* __restrict__ wa_b, const float* __restrict__ wa_ctx,
    float* __restrict__ svec) {
  __shared__ alignas(16) bf16_t Abf[WMP * EDP];      // 28672 B, A matrix (e, later wenc)
  __shared__ float gbuf[WW * 608];                   // 121600 B, gate pre-activations
  __shared__ float wencf[WW * ED];                   // 40000 B, fp32 word encodings
  __shared__ int   tok[WMP];
  __shared__ float scores[WMP];
  __shared__ float alphas[WMP];

  const int sid  = blockIdx.x;          // b*40 + s
  const int tid  = threadIdx.x;
  const int lane = tid & 31;
  const int wave = tid >> 5;

  if (tid < WMP) { tok[tid] = (tid < WW) ? x[sid * WW + tid] : 0; scores[tid] = 0.f; }
  __syncthreads();

  // Gather embeddings into bf16 A layout (zero padding included).
  for (int idx = tid; idx < WMP * EDP; idx += 256) {
    int r = idx / EDP, c = idx % EDP;
    float v = 0.f;
    if (r < WW && c < ED) v = emb[(size_t)tok[r] * ED + c];
    Abf[idx] = (bf16_t)v;
  }
  __syncthreads();

  // ---- gate GEMM: g = e @ [Wih_f ; Wih_b].T   (64x224) x (224x608) ----
  for (int nt = wave; nt < NTG; nt += 8) {
    v8f acc[4] = {};
#pragma unroll
    for (int kt = 0; kt < KT; ++kt) {
      v16bf bf = load_b(Wpack, nt, kt, lane);
#pragma unroll
      for (int mt = 0; mt < 4; ++mt) {
        v16bf af = load_a(Abf, mt, kt, lane);
        acc[mt] = WMMA_BF16(af, bf, acc[mt]);
      }
    }
    int col = nt * 16 + (lane & 15);
    if (col < GN) {
      float bi = (col < 300) ? bih_f[col] : bih_b[col - 300];
#pragma unroll
      for (int mt = 0; mt < 4; ++mt) {
#pragma unroll
        for (int r = 0; r < 8; ++r) {
          int row = mt * 16 + ((lane >> 4) << 3) + r;
          if (row < WW) gbuf[row * 608 + col] = acc[mt][r] + bi;
        }
      }
    }
  }
  __syncthreads();

  // ---- GRU gates from zero hidden state: h = (1-z)*n ----
  for (int idx = tid; idx < WW * ED; idx += 256) {
    int m = idx / ED, c = idx % ED;
    int dir = c / 100, h = c % 100;
    const float* bhh = dir ? bhh_b : bhh_f;
    int base = m * 608 + dir * 300;
    float r = sigf(gbuf[base + h] + bhh[h]);
    float z = sigf(gbuf[base + 100 + h] + bhh[100 + h]);
    float n = tanhf(gbuf[base + 200 + h] + r * bhh[200 + h]);
    wencf[idx] = (1.f - z) * n;
  }
  __syncthreads();
  // Re-use Abf as wenc in bf16 (padding stays zero).
  for (int idx = tid; idx < WW * ED; idx += 256) {
    int m = idx / ED, c = idx % ED;
    Abf[m * EDP + c] = (bf16_t)wencf[idx];
  }
  __syncthreads();

  // ---- attention GEMM: proj = wenc @ Wa.T ; scores[m] += tanh(proj+b)*ctx ----
  for (int nt = wave; nt < NTA; nt += 8) {
    v8f acc[4] = {};
#pragma unroll
    for (int kt = 0; kt < KT; ++kt) {
      v16bf bf = load_b(WApack, nt, kt, lane);
#pragma unroll
      for (int mt = 0; mt < 4; ++mt) {
        v16bf af = load_a(Abf, mt, kt, lane);
        acc[mt] = WMMA_BF16(af, bf, acc[mt]);
      }
    }
    int col = nt * 16 + (lane & 15);
    if (col < AN) {
      float bb = wa_b[col], cc = wa_ctx[col];
#pragma unroll
      for (int mt = 0; mt < 4; ++mt) {
#pragma unroll
        for (int r = 0; r < 8; ++r) {
          int row = mt * 16 + ((lane >> 4) << 3) + r;
          if (row < WW) atomicAdd(&scores[row], tanhf(acc[mt][r] + bb) * cc);
        }
      }
    }
  }
  __syncthreads();

  if (tid == 0) {
    float mx = -1e30f;
    for (int m = 0; m < WW; ++m) mx = fmaxf(mx, scores[m]);
    float s = 0.f;
    for (int m = 0; m < WW; ++m) { float e = __expf(scores[m] - mx); alphas[m] = e; s += e; }
    float inv = 1.f / s;
    for (int m = 0; m < WW; ++m) alphas[m] *= inv;
  }
  __syncthreads();

  for (int d = tid; d < ED; d += 256) {
    float s = 0.f;
    for (int m = 0; m < WW; ++m) s += alphas[m] * wencf[m * ED + d];
    svec[(size_t)sid * ED + d] = s;
  }
}

// ---------------------------------------------------------------------------
// Sentence-level kernel: one block per batch row. Same structure on [40,200],
// plus classifier + log_softmax.
// ---------------------------------------------------------------------------
__global__ __launch_bounds__(256) void sent_kernel(
    const float* __restrict__ svec, const bf16_t* __restrict__ Spack,
    const float* __restrict__ bih_f, const float* __restrict__ bhh_f,
    const float* __restrict__ bih_b, const float* __restrict__ bhh_b,
    const bf16_t* __restrict__ SApack,
    const float* __restrict__ sa_b, const float* __restrict__ sa_ctx,
    const float* __restrict__ fc_W, const float* __restrict__ fc_b,
    float* __restrict__ out) {
  __shared__ alignas(16) bf16_t Abf[SMP * EDP];      // 21504 B
  __shared__ float gbuf[SS * 608];                   // 97280 B
  __shared__ float sencf[SS * ED];                   // 32000 B
  __shared__ float scores[SMP];
  __shared__ float alphas[SMP];
  __shared__ float dvec[ED];
  __shared__ float lg[16];

  const int b    = blockIdx.x;
  const int tid  = threadIdx.x;
  const int lane = tid & 31;
  const int wave = tid >> 5;

  if (tid < SMP) scores[tid] = 0.f;
  for (int idx = tid; idx < SMP * EDP; idx += 256) {
    int r = idx / EDP, c = idx % EDP;
    float v = 0.f;
    if (r < SS && c < ED) v = svec[((size_t)b * SS + r) * ED + c];
    Abf[idx] = (bf16_t)v;
  }
  __syncthreads();

  // ---- gate GEMM (48x224) x (224x608) ----
  for (int nt = wave; nt < NTG; nt += 8) {
    v8f acc[3] = {};
#pragma unroll
    for (int kt = 0; kt < KT; ++kt) {
      v16bf bf = load_b(Spack, nt, kt, lane);
#pragma unroll
      for (int mt = 0; mt < 3; ++mt) {
        v16bf af = load_a(Abf, mt, kt, lane);
        acc[mt] = WMMA_BF16(af, bf, acc[mt]);
      }
    }
    int col = nt * 16 + (lane & 15);
    if (col < GN) {
      float bi = (col < 300) ? bih_f[col] : bih_b[col - 300];
#pragma unroll
      for (int mt = 0; mt < 3; ++mt) {
#pragma unroll
        for (int r = 0; r < 8; ++r) {
          int row = mt * 16 + ((lane >> 4) << 3) + r;
          if (row < SS) gbuf[row * 608 + col] = acc[mt][r] + bi;
        }
      }
    }
  }
  __syncthreads();

  for (int idx = tid; idx < SS * ED; idx += 256) {
    int m = idx / ED, c = idx % ED;
    int dir = c / 100, h = c % 100;
    const float* bhh = dir ? bhh_b : bhh_f;
    int base = m * 608 + dir * 300;
    float r = sigf(gbuf[base + h] + bhh[h]);
    float z = sigf(gbuf[base + 100 + h] + bhh[100 + h]);
    float n = tanhf(gbuf[base + 200 + h] + r * bhh[200 + h]);
    sencf[idx] = (1.f - z) * n;
  }
  __syncthreads();
  for (int idx = tid; idx < SS * ED; idx += 256) {
    int m = idx / ED, c = idx % ED;
    Abf[m * EDP + c] = (bf16_t)sencf[idx];
  }
  __syncthreads();

  // ---- attention GEMM ----
  for (int nt = wave; nt < NTA; nt += 8) {
    v8f acc[3] = {};
#pragma unroll
    for (int kt = 0; kt < KT; ++kt) {
      v16bf bf = load_b(SApack, nt, kt, lane);
#pragma unroll
      for (int mt = 0; mt < 3; ++mt) {
        v16bf af = load_a(Abf, mt, kt, lane);
        acc[mt] = WMMA_BF16(af, bf, acc[mt]);
      }
    }
    int col = nt * 16 + (lane & 15);
    if (col < AN) {
      float bb = sa_b[col], cc = sa_ctx[col];
#pragma unroll
      for (int mt = 0; mt < 3; ++mt) {
#pragma unroll
        for (int r = 0; r < 8; ++r) {
          int row = mt * 16 + ((lane >> 4) << 3) + r;
          if (row < SS) atomicAdd(&scores[row], tanhf(acc[mt][r] + bb) * cc);
        }
      }
    }
  }
  __syncthreads();

  if (tid == 0) {
    float mx = -1e30f;
    for (int m = 0; m < SS; ++m) mx = fmaxf(mx, scores[m]);
    float s = 0.f;
    for (int m = 0; m < SS; ++m) { float e = __expf(scores[m] - mx); alphas[m] = e; s += e; }
    float inv = 1.f / s;
    for (int m = 0; m < SS; ++m) alphas[m] *= inv;
  }
  __syncthreads();

  for (int d = tid; d < ED; d += 256) {
    float s = 0.f;
    for (int m = 0; m < SS; ++m) s += alphas[m] * sencf[m * ED + d];
    dvec[d] = s;
  }
  __syncthreads();

  if (tid < NC) {
    float s = fc_b[tid];
    for (int d = 0; d < ED; ++d) s += dvec[d] * fc_W[tid * ED + d];
    lg[tid] = s;
  }
  __syncthreads();
  if (tid == 0) {
    float mx = -1e30f;
    for (int c = 0; c < NC; ++c) mx = fmaxf(mx, lg[c]);
    float s = 0.f;
    for (int c = 0; c < NC; ++c) s += __expf(lg[c] - mx);
    float lse = mx + logf(s);
    for (int c = 0; c < NC; ++c) out[b * NC + c] = lg[c] - lse;
  }
}

// ---------------------------------------------------------------------------
extern "C" void kernel_launch(void* const* d_in, const int* in_sizes, int n_in,
                              void* d_out, int out_size, void* d_ws, size_t ws_size,
                              hipStream_t stream) {
  const int*   x      = (const int*)d_in[0];
  const float* emb    = (const float*)d_in[1];
  const float* wWf    = (const float*)d_in[2];
  const float* wbih_f = (const float*)d_in[3];
  const float* wbhh_f = (const float*)d_in[4];
  const float* wWb    = (const float*)d_in[5];
  const float* wbih_b = (const float*)d_in[6];
  const float* wbhh_b = (const float*)d_in[7];
  const float* waW    = (const float*)d_in[8];
  const float* wab    = (const float*)d_in[9];
  const float* wactx  = (const float*)d_in[10];
  const float* sWf    = (const float*)d_in[11];
  const float* sbih_f = (const float*)d_in[12];
  const float* sbhh_f = (const float*)d_in[13];
  const float* sWb    = (const float*)d_in[14];
  const float* sbih_b = (const float*)d_in[15];
  const float* sbhh_b = (const float*)d_in[16];
  const float* saW    = (const float*)d_in[17];
  const float* sab    = (const float*)d_in[18];
  const float* sactx  = (const float*)d_in[19];
  const float* fcW    = (const float*)d_in[20];
  const float* fcb    = (const float*)d_in[21];
  float* out = (float*)d_out;

  // Workspace layout: svec f32 [128*40*200], then 4 packed bf16 weight buffers.
  float*  svec   = (float*)d_ws;
  bf16_t* wordWp = (bf16_t*)(svec + (size_t)BB * SS * ED);
  bf16_t* waWp   = wordWp + (size_t)NTG * KT * FRAG;
  bf16_t* sentWp = waWp   + (size_t)NTA * KT * FRAG;
  bf16_t* saWp   = sentWp + (size_t)NTG * KT * FRAG;

  const int totG = NTG * KT * FRAG;   // 136192
  const int totA = NTA * KT * FRAG;   // 46592
  pack_b_kernel<<<(totG + 255) / 256, 256, 0, stream>>>(wWf, wWb, 300, GN, NTG, wordWp);
  pack_b_kernel<<<(totA + 255) / 256, 256, 0, stream>>>(waW, waW, 200, AN, NTA, waWp);
  pack_b_kernel<<<(totG + 255) / 256, 256, 0, stream>>>(sWf, sWb, 300, GN, NTG, sentWp);
  pack_b_kernel<<<(totA + 255) / 256, 256, 0, stream>>>(saW, saW, 200, AN, NTA, saWp);

  word_kernel<<<BB * SS, 256, 0, stream>>>(x, emb, wbih_f, wbhh_f, wbih_b, wbhh_b,
                                           wordWp, waWp, wab, wactx, svec);
  sent_kernel<<<BB, 256, 0, stream>>>(svec, sentWp, sbih_f, sbhh_f, sbih_b, sbhh_b,
                                      saWp, sab, sactx, fcW, fcb, out);
}
